// TemporalAbstraction_39685497815925
// MI455X (gfx1250) — matrix operand order repaired and checked by reading, not compiled, not run on hardware
//
#include <hip/hip_runtime.h>
#include <hip/hip_bf16.h>
#include <math.h>

// ---------------------------------------------------------------------------
// 3-layer GRU (B=16, L=512, D=1024) for gfx1250.
//  - bf16 WMMA (v_wmma_f32_16x16x32_bf16) for the batched input projection
//    and the per-step recurrent GEMM, fp32 accumulation.
//  - Persistent scan kernel with grid-wide barrier. Each workgroup stages its
//    64-column W_hh slice (128 KB bf16) into LDS ONCE via the Tensor Data
//    Mover (tensor_load_to_lds + s_wait_tensorcnt), then serves all 512
//    steps' B-fragments from LDS -> no per-step L2 weight traffic on the
//    sequential critical path.
// ---------------------------------------------------------------------------

typedef __attribute__((ext_vector_type(16))) __bf16 v16bf;
typedef __attribute__((ext_vector_type(8)))  __bf16 v8bf;
typedef __attribute__((ext_vector_type(8)))  float  v8f;
typedef __attribute__((ext_vector_type(4))) unsigned int u32x4;
typedef __attribute__((ext_vector_type(4))) int          i32x4;
typedef __attribute__((ext_vector_type(8))) int          i32x8;

#if defined(__has_builtin)
#if __has_builtin(__builtin_amdgcn_tensor_load_to_lds)
#define GRU_HAVE_TDM 1
#endif
#endif
#ifndef GRU_HAVE_TDM
#define GRU_HAVE_TDM 0
#endif

#define DD    1024
#define LL    512
#define BB    16
#define G3    3072                 // 3*D
#define MROWS (BB * LL)            // 8192
#define KSTEPS (DD / 32)           // 32 wmma K-steps

#define SCAN_BLOCKS  48
#define SCAN_THREADS 128
#define SCAN_WAVES   (SCAN_BLOCKS * SCAN_THREADS / 32)   // 192 == G3/16
static_assert(SCAN_WAVES * 16 == G3, "scan waves must cover N");

// W_hh LDS staging: 64 columns/WG, row = 1024 bf16 = 2048 B of data,
// TDM pad inserts 16 B after every 1024 B of data -> 2080 B row stride.
#define WROW_DATA   2048
#define WROW_STRIDE 2080
#define WLDS_BYTES  (64 * WROW_STRIDE)   // 133120

union AFrag { v16bf v; v8bf h[2]; };

// ---------------------------------------------------------------------------
// Grid-wide barrier (counter + generation, agent scope).
// ---------------------------------------------------------------------------
__device__ __forceinline__ void grid_sync(unsigned* cnt, unsigned* gen) {
  __threadfence();
  __syncthreads();
  if (threadIdx.x == 0) {
    unsigned g = __hip_atomic_load(gen, __ATOMIC_ACQUIRE, __HIP_MEMORY_SCOPE_AGENT);
    unsigned arrived =
        __hip_atomic_fetch_add(cnt, 1u, __ATOMIC_ACQ_REL, __HIP_MEMORY_SCOPE_AGENT);
    if (arrived == gridDim.x - 1) {
      __hip_atomic_store(cnt, 0u, __ATOMIC_RELAXED, __HIP_MEMORY_SCOPE_AGENT);
      __hip_atomic_fetch_add(gen, 1u, __ATOMIC_ACQ_REL, __HIP_MEMORY_SCOPE_AGENT);
    } else {
      while (__hip_atomic_load(gen, __ATOMIC_ACQUIRE, __HIP_MEMORY_SCOPE_AGENT) == g)
        __builtin_amdgcn_s_sleep(1);
    }
  }
  __syncthreads();
  __threadfence();
}

// ---------------------------------------------------------------------------
// fp32 -> bf16 convert (grid-stride)
// ---------------------------------------------------------------------------
__global__ void f32_to_bf16_kernel(const float* __restrict__ in,
                                   __bf16* __restrict__ out, int n) {
  int i = blockIdx.x * blockDim.x + threadIdx.x;
  int stride = gridDim.x * blockDim.x;
  for (; i < n; i += stride) out[i] = (__bf16)in[i];
}

// ---------------------------------------------------------------------------
// Per-layer scan state init: h = 0 (fp32 + bf16), barrier words = 0.
// ---------------------------------------------------------------------------
__global__ void scan_init_kernel(float* h_f32, __bf16* h_bf, unsigned* bar) {
  int i = blockIdx.x * blockDim.x + threadIdx.x;
  int stride = gridDim.x * blockDim.x;
  for (; i < BB * DD; i += stride) {
    h_f32[i] = 0.0f;
    h_bf[i] = (__bf16)0.0f;
  }
  if (blockIdx.x == 0 && threadIdx.x == 0) { bar[0] = 0u; bar[32] = 0u; }
}

// ---------------------------------------------------------------------------
// xp[8192,3072] = Xbf[8192,1024] @ W_ih^T + b_ih   (bf16 WMMA, f32 acc)
// Per-wave 16(M) x 64(N) tile; A-fragment reused 4x.
//   A: lane(m=lane&15, h=lane>>4): 16 elems at K0+16h of row m.
//   B: lane(n=lane&15, h):        16 elems at K0+16h of W row n (B=W^T).
//   C: lane n = lane&15, vgpr r -> row r + 8*(lane>>4).
// ---------------------------------------------------------------------------
__global__ void __launch_bounds__(128)
gemm_xp_kernel(const __bf16* __restrict__ A, const __bf16* __restrict__ W,
               const float* __restrict__ bias, float* __restrict__ C) {
  const int lane = threadIdx.x & 31;
  const int wave = blockIdx.x * 4 + (threadIdx.x >> 5);
  const int mt = wave / 48;              // 512 M-tiles
  const int nb = wave % 48;              // 48 N-blocks of 64
  const int nlo = lane & 15;
  const int half = lane >> 4;

  const __bf16* arow = A + (size_t)(mt * 16 + nlo) * DD;
  const int n0 = nb * 64 + nlo;
  const __bf16* w0 = W + (size_t)(n0 + 0)  * DD + 16 * half;
  const __bf16* w1 = W + (size_t)(n0 + 16) * DD + 16 * half;
  const __bf16* w2 = W + (size_t)(n0 + 32) * DD + 16 * half;
  const __bf16* w3 = W + (size_t)(n0 + 48) * DD + 16 * half;

  v8f acc0 = {0,0,0,0,0,0,0,0};
  v8f acc1 = acc0, acc2 = acc0, acc3 = acc0;

#pragma unroll 4
  for (int kk = 0; kk < KSTEPS; ++kk) {
    const int K0 = kk * 32;
    AFrag a;
    a.h[0] = *(const v8bf*)(arow + K0 + 16 * half);
    a.h[1] = *(const v8bf*)(arow + K0 + 16 * half + 8);
    v16bf b0 = *(const v16bf*)(w0 + K0);
    v16bf b1 = *(const v16bf*)(w1 + K0);
    v16bf b2 = *(const v16bf*)(w2 + K0);
    v16bf b3 = *(const v16bf*)(w3 + K0);
    acc0 = __builtin_amdgcn_wmma_f32_16x16x32_bf16(false, a.v, false, b0, (short)0, acc0, false, false);
    acc1 = __builtin_amdgcn_wmma_f32_16x16x32_bf16(false, a.v, false, b1, (short)0, acc1, false, false);
    acc2 = __builtin_amdgcn_wmma_f32_16x16x32_bf16(false, a.v, false, b2, (short)0, acc2, false, false);
    acc3 = __builtin_amdgcn_wmma_f32_16x16x32_bf16(false, a.v, false, b3, (short)0, acc3, false, false);
  }

  const int mbase = mt * 16 + 8 * half;
#pragma unroll
  for (int s = 0; s < 4; ++s) {
    v8f acc = (s == 0) ? acc0 : (s == 1) ? acc1 : (s == 2) ? acc2 : acc3;
    const int col = nb * 64 + s * 16 + nlo;
    const float bv = bias[col];
#pragma unroll
    for (int r = 0; r < 8; ++r)
      C[(size_t)(mbase + r) * G3 + col] = acc[r] + bv;
  }
}

// ---------------------------------------------------------------------------
// Persistent GRU scan for one layer. 48 blocks x 128 threads = 192 waves,
// wave w owns output columns [16w, 16w+16) for all 512 steps.
// W_hh slice staged to LDS once via TDM; B-fragments come from LDS.
// ---------------------------------------------------------------------------
__global__ void __launch_bounds__(SCAN_THREADS)
gru_scan_kernel(const float* __restrict__ xp, const __bf16* __restrict__ Whh,
                const float* __restrict__ bhh,
                float* h_f32, __bf16* h_bf, float* gh,
                __bf16* out_bf, float* out_f32, float* state_out,
                unsigned* bar) {
  extern __shared__ char wlds[];   // 64 rows x 2080 B (TDM-padded W slice)

  const int lane = threadIdx.x & 31;
  const int wwg  = threadIdx.x >> 5;                         // wave in WG 0..3
  const int nlo  = lane & 15;
  const int half = lane >> 4;
  const int ncol = blockIdx.x * 64 + wwg * 16 + nlo;         // 0..3071
  const int lcol = wwg * 16 + nlo;                           // 0..63 (LDS col)
  const __bf16* arow = h_bf + (size_t)nlo * DD + 16 * half;  // A row m = nlo
  const int tid = blockIdx.x * SCAN_THREADS + threadIdx.x;   // 0..6143
  unsigned* cnt = bar;
  unsigned* gen = bar + 32;

  // ---- stage this WG's 64 W_hh columns into LDS (once per layer) ----------
  const unsigned long long gaddr =
      (unsigned long long)(const char*)Whh +
      (unsigned long long)blockIdx.x * (64ull * DD * 2ull);
#if GRU_HAVE_TDM
  if (wwg == 0) {
    // D# group0: count=1 | lds_addr=0 | global_addr | type=2
    u32x4 g0 = { 1u,
                 0u,
                 (unsigned)(gaddr & 0xffffffffull),
                 (unsigned)((gaddr >> 32) & 0x01ffffffull) | (2u << 30) };
    // D# group1: data_size=2B, pad_enable, pad_interval=1KB, pad_amount=16B,
    // tensor_dim0=1024, tensor_dim1=64, tile_dim0=1024, tile_dim1=64,
    // tensor_dim0_stride=1024.
    i32x8 g1 = { (int)0x07D10000u,
                 (int)((1024u & 0xffffu) << 16),
                 (int)((64u   & 0xffffu) << 16),
                 (int)(1024u << 16),
                 (int)64,
                 (int)1024,
                 0, 0 };
    i32x4 gz = {0, 0, 0, 0};
#if __clang_major__ >= 23
    i32x8 gz8 = {0, 0, 0, 0, 0, 0, 0, 0};
    __builtin_amdgcn_tensor_load_to_lds(g0, g1, gz, gz, gz8, 0);
#else
    __builtin_amdgcn_tensor_load_to_lds(g0, g1, gz, gz, 0);
#endif
    __builtin_amdgcn_s_wait_tensorcnt(0);
  }
#else
  {
    const unsigned* wsrc = (const unsigned*)gaddr;
    for (int idx = threadIdx.x; idx < 64 * (WROW_DATA / 4); idx += SCAN_THREADS) {
      const int row = idx >> 9;                 // /512 dwords per row
      const int cd  = idx & 511;
      *(unsigned*)(wlds + row * WROW_STRIDE + cd * 4 + (cd >= 256 ? 16 : 0)) =
          wsrc[idx];
    }
  }
#endif
  __syncthreads();

  const char* wbase = wlds + lcol * WROW_STRIDE + 32 * half;

  for (int t = 0; t < LL; ++t) {
    // ---- recurrent GEMM: gh[16,3072] = h @ W_hh^T (B from LDS) -----------
    v8f acc = {0,0,0,0,0,0,0,0};
#pragma unroll 8
    for (int kk = 0; kk < 16; ++kk) {          // data offset < 1024 B: no pad
      const int K0 = kk * 32;
      AFrag a, b;
      a.h[0] = *(const v8bf*)(arow + K0);
      a.h[1] = *(const v8bf*)(arow + K0 + 8);
      b.h[0] = *(const v8bf*)(wbase + kk * 64);
      b.h[1] = *(const v8bf*)(wbase + kk * 64 + 16);
      acc = __builtin_amdgcn_wmma_f32_16x16x32_bf16(false, a.v, false, b.v, (short)0, acc, false, false);
    }
#pragma unroll 8
    for (int kk = 16; kk < 32; ++kk) {         // data offset >= 1024 B: +16 pad
      const int K0 = kk * 32;
      AFrag a, b;
      a.h[0] = *(const v8bf*)(arow + K0);
      a.h[1] = *(const v8bf*)(arow + K0 + 8);
      b.h[0] = *(const v8bf*)(wbase + kk * 64 + 16);
      b.h[1] = *(const v8bf*)(wbase + kk * 64 + 32);
      acc = __builtin_amdgcn_wmma_f32_16x16x32_bf16(false, a.v, false, b.v, (short)0, acc, false, false);
    }
#pragma unroll
    for (int r = 0; r < 8; ++r)
      gh[(size_t)(r + 8 * half) * G3 + ncol] = acc[r];

    grid_sync(cnt, gen);

    // ---- gate math + state update: 16*1024 elems over 6144 threads --------
    for (int idx = tid; idx < BB * DD; idx += SCAN_BLOCKS * SCAN_THREADS) {
      const int m = idx >> 10;            // batch
      const int j = idx & (DD - 1);       // feature
      const size_t row = (size_t)m * LL + t;
      const float xr = xp[row * G3 + j];
      const float xz = xp[row * G3 + DD + j];
      const float xn = xp[row * G3 + 2 * DD + j];
      const float hr = gh[(size_t)m * G3 + j] + bhh[j];
      const float hz = gh[(size_t)m * G3 + DD + j] + bhh[DD + j];
      const float hn = gh[(size_t)m * G3 + 2 * DD + j] + bhh[2 * DD + j];
      const float rg = 1.0f / (1.0f + __expf(-(xr + hr)));
      const float zg = 1.0f / (1.0f + __expf(-(xz + hz)));
      const float ng = tanhf(xn + rg * hn);
      const float hnew = (1.0f - zg) * ng + zg * h_f32[idx];
      h_f32[idx] = hnew;
      h_bf[idx] = (__bf16)hnew;
      out_bf[row * DD + j] = (__bf16)hnew;          // next layer's input
      if (out_f32) out_f32[row * DD + j] = hnew;    // final layer -> d_out
      if (t == LL - 1) state_out[idx] = hnew;       // temporal state
    }

    grid_sync(cnt, gen);
  }
}

// ---------------------------------------------------------------------------
// Workspace layout (bytes)
// ---------------------------------------------------------------------------
#define OFF_XP   ((size_t)0)                                   // 100,663,296
#define OFF_X0   (OFF_XP + (size_t)MROWS * G3 * 4)             //  16,777,216
#define OFF_X1   (OFF_X0 + (size_t)MROWS * DD * 2)             //  16,777,216
#define OFF_WIH  (OFF_X1 + (size_t)MROWS * DD * 2)             //   6,291,456
#define OFF_WHH  (OFF_WIH + (size_t)G3 * DD * 2)               //   6,291,456
#define OFF_GH   (OFF_WHH + (size_t)G3 * DD * 2)               //     196,608
#define OFF_HF   (OFF_GH + (size_t)BB * G3 * 4)                //      65,536
#define OFF_HB   (OFF_HF + (size_t)BB * DD * 4)                //      32,768
#define OFF_BAR  (OFF_HB + (size_t)BB * DD * 2)                //         256

extern "C" void kernel_launch(void* const* d_in, const int* in_sizes, int n_in,
                              void* d_out, int out_size, void* d_ws, size_t ws_size,
                              hipStream_t stream) {
  const float* x    = (const float*)d_in[0];   // [16,512,1024]
  const float* W_ih = (const float*)d_in[1];   // [3,3072,1024]
  const float* W_hh = (const float*)d_in[2];   // [3,3072,1024]
  const float* b_ih = (const float*)d_in[3];   // [3,3072]
  const float* b_hh = (const float*)d_in[4];   // [3,3072]
  float* out = (float*)d_out;                  // [16,512,1024] + 3*[16,1024]

  char* ws = (char*)d_ws;
  float*    xp     = (float*)(ws + OFF_XP);
  __bf16*   xbf[2] = {(__bf16*)(ws + OFF_X0), (__bf16*)(ws + OFF_X1)};
  __bf16*   Wihb   = (__bf16*)(ws + OFF_WIH);
  __bf16*   Whhb   = (__bf16*)(ws + OFF_WHH);
  float*    gh     = (float*)(ws + OFF_GH);
  float*    h_f32  = (float*)(ws + OFF_HF);
  __bf16*   h_bf   = (__bf16*)(ws + OFF_HB);
  unsigned* bar    = (unsigned*)(ws + OFF_BAR);

  // layer-0 input in bf16
  f32_to_bf16_kernel<<<2048, 256, 0, stream>>>(x, xbf[0], MROWS * DD);

  int cur = 0;
  for (int layer = 0; layer < 3; ++layer) {
    const size_t woff = (size_t)layer * G3 * DD;
    f32_to_bf16_kernel<<<2048, 256, 0, stream>>>(W_ih + woff, Wihb, G3 * DD);
    f32_to_bf16_kernel<<<2048, 256, 0, stream>>>(W_hh + woff, Whhb, G3 * DD);

    // xp = x_layer @ W_ih^T + b_ih  (8192x3072x1024 bf16 WMMA GEMM)
    gemm_xp_kernel<<<6144, 128, 0, stream>>>(xbf[cur], Wihb,
                                             b_ih + (size_t)layer * G3, xp);

    scan_init_kernel<<<64, 256, 0, stream>>>(h_f32, h_bf, bar);

    gru_scan_kernel<<<SCAN_BLOCKS, SCAN_THREADS, WLDS_BYTES, stream>>>(
        xp, Whhb, b_hh + (size_t)layer * G3,
        h_f32, h_bf, gh,
        xbf[1 - cur],
        (layer == 2) ? out : (float*)nullptr,
        out + (size_t)MROWS * DD + (size_t)layer * BB * DD,
        bar);

    cur = 1 - cur;
  }
}
